// W4Linear_43336220016659
// MI455X (gfx1250) — matrix operand order repaired and checked by reading, compiled, not run on hardware
//
#include <hip/hip_runtime.h>
#include <stdint.h>

typedef __attribute__((ext_vector_type(16))) _Float16 v16h;
typedef __attribute__((ext_vector_type(8)))  _Float16 v8h;
typedef __attribute__((ext_vector_type(2)))  _Float16 h2;
typedef __attribute__((ext_vector_type(8)))  float    v8f;
typedef __attribute__((ext_vector_type(4)))  uint32_t v4u;
typedef __attribute__((ext_vector_type(4)))  int      v4i;

#define GLOBAL_AS __attribute__((address_space(1)))
#define LDS_AS    __attribute__((address_space(3)))

#define BM 128
#define BN 256
#define BK 32
#define LDP 40   // padded LDS row stride (halves): 80B rows -> conflict-free, 16B-aligned b128

#if defined(__has_builtin)
#if __has_builtin(__builtin_amdgcn_global_load_async_to_lds_b128) && \
    __has_builtin(__builtin_amdgcn_s_wait_asynccnt)
#define USE_ASYNC_LDS 1
#endif
#endif

__global__ __launch_bounds__(256) void w4_gemm_wmma(
    const _Float16* __restrict__ x,     // [M,K] fp16 row-major
    const int8_t*   __restrict__ qw,    // [N,K] int8 (int4 range) row-major
    const _Float16* __restrict__ scale, // [N]
    const _Float16* __restrict__ bias,  // [N]
    _Float16*       __restrict__ y,     // [M,N] fp16 row-major
    int M, int N, int K)
{
    __shared__ __align__(16) _Float16 a_lds[2][BM * LDP];   // 2 x 10 KB
    __shared__ __align__(16) _Float16 b_lds[2][BN * LDP];   // 2 x 20 KB

    const int tid  = threadIdx.x;
    const int lane = tid & 31;
    const int wave = tid >> 5;
    const int wm   = wave >> 2;   // 0..1 : 64-row strip
    const int wn   = wave & 3;    // 0..3 : 64-col strip

    const int block_m = blockIdx.y * BM;
    const int block_n = blockIdx.x * BN;

    const int lr = lane & 15;     // row (A) / col (B,C) within 16
    const int hk = lane >> 4;     // K-half selector per WMMA VGPR layout

    // ---- staging helpers ----
    auto stageA = [&](int ks, int buf) {
#pragma unroll
        for (int j = 0; j < 2; ++j) {
            const int c   = tid + 256 * j;   // 512 chunks of 16B (128 rows x 64B)
            const int row = c >> 2;
            const int ko  = (c & 3) * 8;
            const _Float16* src = x + (size_t)(block_m + row) * K + ks * BK + ko;
            _Float16*       dst = &a_lds[buf][row * LDP + ko];
#ifdef USE_ASYNC_LDS
            __builtin_amdgcn_global_load_async_to_lds_b128(
                (GLOBAL_AS v4i*)src, (LDS_AS v4i*)dst, 0, 0);
#else
            *(v8h*)dst = *(const v8h*)src;
#endif
        }
    };
    auto loadB = [&](int ks, v4u raw[2]) {
#pragma unroll
        for (int j = 0; j < 2; ++j) {
            const int c    = tid + 256 * j;  // 512 chunks of 16B (256 rows x 32B)
            const int nrow = c >> 1;
            const int kc   = c & 1;
            const int8_t* bp = qw + (size_t)(block_n + nrow) * K + ks * BK + kc * 16;
            raw[j] = *(const v4u*)bp;
            __builtin_prefetch(bp + 2 * BK, 0, 1);   // gfx1250 global_prefetch_b8
        }
    };
    auto storeB = [&](const v4u raw[2], int buf) {
        const h2 magic = { (_Float16)-1032.0f, (_Float16)-1032.0f };
#pragma unroll
        for (int j = 0; j < 2; ++j) {
            const int c    = tid + 256 * j;
            const int nrow = c >> 1;
            const int kc   = c & 1;
            v4u outA, outB;
#pragma unroll
            for (int w = 0; w < 4; ++w) {
                // magic dequant: (v&0xF)^0x8 == v+8 ; 0x6400|(v+8) is fp16(1032+v)
                const uint32_t u = (raw[j][w] & 0x0F0F0F0Fu) ^ 0x08080808u;
                uint32_t p0 = __builtin_amdgcn_perm(0x64646464u, u, 0x05010400u);
                uint32_t p1 = __builtin_amdgcn_perm(0x64646464u, u, 0x07030602u);
                p0 = __builtin_bit_cast(uint32_t, h2(__builtin_bit_cast(h2, p0) + magic));
                p1 = __builtin_bit_cast(uint32_t, h2(__builtin_bit_cast(h2, p1) + magic));
                if (w < 2) { outA[2*w]   = p0; outA[2*w+1] = p1; }
                else       { outB[2*w-4] = p0; outB[2*w-3] = p1; }
            }
            *(v4u*)(&b_lds[buf][nrow * LDP + kc * 16])     = outA;
            *(v4u*)(&b_lds[buf][nrow * LDP + kc * 16 + 8]) = outB;
        }
    };

    v8f acc[4][4];
#pragma unroll
    for (int i = 0; i < 4; ++i)
#pragma unroll
        for (int j = 0; j < 4; ++j)
            acc[i][j] = (v8f)0.0f;

    const int nsteps = K / BK;

    // ---- prologue: fill buffer 0 ----
    v4u rawB[2];
    stageA(0, 0);
    loadB(0, rawB);
    storeB(rawB, 0);
#ifdef USE_ASYNC_LDS
    __builtin_amdgcn_s_wait_asynccnt(0);
#endif
    __syncthreads();

    for (int ks = 0; ks < nsteps; ++ks) {
        const int cur = ks & 1;
        const bool has_next = (ks + 1) < nsteps;

        // issue next tile's global traffic before compute (latency hidden by WMMAs)
        if (has_next) {
            stageA(ks + 1, cur ^ 1);
            loadB(ks + 1, rawB);
        }

        // ---- A fragments (ISA 16-bit A layout: lane=M row, hk selects K-half) ----
        v16h afrag[4];
#pragma unroll
        for (int mt = 0; mt < 4; ++mt) {
            const int r = wm * 64 + mt * 16 + lr;
            const v8h lo = *(const v8h*)(&a_lds[cur][r * LDP + hk * 8]);
            const v8h hi = *(const v8h*)(&a_lds[cur][r * LDP + 16 + hk * 8]);
            afrag[mt] = __builtin_shufflevector(lo, hi,
                0,1,2,3,4,5,6,7,8,9,10,11,12,13,14,15);
        }

        // ---- B fragments + 16 WMMAs (each fragment reused 4x) ----
#pragma unroll
        for (int nt = 0; nt < 4; ++nt) {
            const int nl = wn * 64 + nt * 16 + lr;
            const v8h blo = *(const v8h*)(&b_lds[cur][nl * LDP + hk * 8]);
            const v8h bhi = *(const v8h*)(&b_lds[cur][nl * LDP + 16 + hk * 8]);
            const v16h bfrag = __builtin_shufflevector(blo, bhi,
                0,1,2,3,4,5,6,7,8,9,10,11,12,13,14,15);
#pragma unroll
            for (int mt = 0; mt < 4; ++mt)
                acc[mt][nt] = __builtin_amdgcn_wmma_f32_16x16x32_f16(
                    false, afrag[mt], false, bfrag, (short)0, acc[mt][nt], false, false);
        }

        // dequant + store next B tile after WMMAs are issued
        if (has_next) {
            storeB(rawB, cur ^ 1);
#ifdef USE_ASYNC_LDS
            __builtin_amdgcn_s_wait_asynccnt(0);
#endif
        }
        __syncthreads();   // single barrier per K-step (double-buffered)
    }

    // ---- epilogue: fused per-channel scale + bias in f32, store fp16 ----
#pragma unroll
    for (int nt = 0; nt < 4; ++nt) {
        const int n = block_n + wn * 64 + nt * 16 + lr;
        const float s = (float)scale[n];
        const float b = (float)bias[n];
#pragma unroll
        for (int mt = 0; mt < 4; ++mt) {
            const int mbase = block_m + wm * 64 + mt * 16 + hk * 8; // C layout: hi lanes -> M+8
#pragma unroll
            for (int i = 0; i < 8; ++i) {
                const float v = acc[mt][nt][i] * s + b;
                y[(size_t)(mbase + i) * N + n] = (_Float16)v;
            }
        }
    }
}

extern "C" void kernel_launch(void* const* d_in, const int* in_sizes, int n_in,
                              void* d_out, int out_size, void* d_ws, size_t ws_size,
                              hipStream_t stream) {
    const _Float16* x     = (const _Float16*)d_in[0];
    const int8_t*   qw    = (const int8_t*)  d_in[1];
    const _Float16* scale = (const _Float16*)d_in[2];
    const _Float16* bias  = (const _Float16*)d_in[3];
    _Float16*       y     = (_Float16*)d_out;

    const int N = in_sizes[2];                  // OUT = 11008
    const int K = in_sizes[1] / N;              // IN  = 4096
    const int M = in_sizes[0] / K;              // B*S = 8192

    dim3 grid(N / BN, M / BM);                  // 43 x 64 workgroups
    w4_gemm_wmma<<<grid, 256, 0, stream>>>(x, qw, scale, bias, y, M, N, K);
}
